// ConstraintMoE_85237920956536
// MI455X (gfx1250) — matrix operand order repaired and checked by reading, not compiled, not run on hardware
//
#include <hip/hip_runtime.h>

// ---------------------------------------------------------------------------
// ConstraintMoE for MI455X (gfx1250): bf16 WMMA GEMM pipeline.
// Weights pre-converted to bf16 per expert; GEMM tiles (A and B) move via
// CDNA5 async global->LDS copies (ASYNCcnt), double-buffered LDS.
// E=6 experts; per expert: x1=act(h@W1+b1), x2=act(x1@W2+b2),
// main += x2@Wm+bm, aux[e] = x2@Wc+bc.
// Shapes: BT=8192, D=1024, H=4096, OUT=4.
// ---------------------------------------------------------------------------

typedef __attribute__((ext_vector_type(16))) __bf16 bf16x16;
typedef __attribute__((ext_vector_type(8)))  float  f32x8;
typedef __attribute__((ext_vector_type(4)))  float  f32x4;
typedef __attribute__((ext_vector_type(4)))  unsigned int u32x4;
typedef __attribute__((ext_vector_type(2)))  unsigned int u32x2;

union ABfrag { bf16x16 v; u32x4 u[2]; };

// float -> bf16 round-to-nearest-even (packed pair)
__device__ inline unsigned pack2bf(float a, float b) {
    unsigned ua = __float_as_uint(a), ub = __float_as_uint(b);
    ua += 0x7FFFu + ((ua >> 16) & 1u);
    ub += 0x7FFFu + ((ub >> 16) & 1u);
    return (ua >> 16) | (ub & 0xFFFF0000u);
}
__device__ inline unsigned short f2bf(float f) {
    unsigned u = __float_as_uint(f);
    u += 0x7FFFu + ((u >> 16) & 1u);
    return (unsigned short)(u >> 16);
}
__device__ inline float bf2f(unsigned short h) {
    return __uint_as_float(((unsigned)h) << 16);
}

// ---------------------------------------------------------------------------
// CDNA5 async global->LDS copy (16B per lane), tracked with ASYNCcnt.
// LDS destination: flat-pointer low 32 bits == wave-relative LDS byte offset
// (ISA 10.2: LDS aperture keeps addr[31:0]).
// ---------------------------------------------------------------------------
__device__ inline void async_ld_b128(void* lds_dst, const void* gsrc) {
    unsigned loff = (unsigned)(size_t)lds_dst;
    asm volatile("global_load_async_to_lds_b128 %0, %1, off"
                 :: "v"(loff), "v"(gsrc)
                 : "memory");
}
__device__ inline void wait_asynccnt0() {
    asm volatile("s_wait_asynccnt 0x0" ::: "memory");
}

// ACT: 0=relu, 1=gelu(tanh approx, jax default), 2=tanh, 3=silu, 4=identity
template<int ACT>
__device__ inline float apply_act(float x) {
    if (ACT == 0) return fmaxf(x, 0.0f);
    if (ACT == 1) {
        float x3 = x * x * x;
        return 0.5f * x * (1.0f + tanhf(0.7978845608028654f * (x + 0.044715f * x3)));
    }
    if (ACT == 2) return tanhf(x);
    if (ACT == 3) return x * (1.0f / (1.0f + __expf(-x)));
    return x;
}

// ---------------------------------------------------------------------------
// GEMM: Out = act(A[M,K](bf16) @ Wb[K,N](bf16) + bias[N])
// MODE 0: Out is bf16 [M,N] (activation applied)
// MODE 1: Out is f32  [M,N]; initOut ? store : accumulate (ACT=4 identity)
// Block: 256 threads, 128x128 tile; wave = 64x32 (4x2 WMMA 16x16x32 tiles).
// Double-buffered LDS; ALL tile traffic via async global->LDS (ASYNCcnt).
// M,N multiples of 128; K multiple of 32.
// ---------------------------------------------------------------------------
template<int ACT, int MODE>
__global__ __launch_bounds__(256)
void gemm_bias_act(const unsigned short* __restrict__ A,
                   const unsigned short* __restrict__ Wb,
                   const float* __restrict__ bias,
                   void* __restrict__ Out,
                   int M, int N, int K, int initOut)
{
    __shared__ unsigned short sA[2][128 * 32];   // [buf][row][k]  2 x 8 KB
    __shared__ unsigned short sB[2][32 * 128];   // [buf][k][col]  2 x 8 KB

    const int tid  = threadIdx.x;
    const int lane = tid & 31;
    const int wid  = tid >> 5;
    const int bm   = blockIdx.y * 128;
    const int bn   = blockIdx.x * 128;
    const int wm   = (wid >> 2) * 64;   // wave row offset in block tile
    const int wn   = (wid & 3) * 32;    // wave col offset in block tile

    f32x8 acc[4][2];
#pragma unroll
    for (int i = 0; i < 4; ++i)
#pragma unroll
        for (int j = 0; j < 2; ++j) acc[i][j] = 0.0f;

    // ---- tile loaders: pure async copies, no VGPR round-trip --------------
    auto loadA = [&](int buf, int k0) {      // 128x32 bf16: 2 async b128/thread
#pragma unroll
        for (int it = 0; it < 2; ++it) {
            int i = tid + it * 256;
            int r = i >> 2, cc = (i & 3) << 3;
            async_ld_b128(&sA[buf][r * 32 + cc],
                          &A[(size_t)(bm + r) * K + k0 + cc]);
        }
    };
    auto loadB = [&](int buf, int k0) {      // 32x128 bf16: 2 async b128/thread
#pragma unroll
        for (int it = 0; it < 2; ++it) {
            int i = tid + it * 256;
            int r = i >> 4, cc = (i & 15) << 3;
            const unsigned short* wp = &Wb[(size_t)(k0 + r) * N + bn + cc];
            async_ld_b128(&sB[buf][r * 128 + cc], wp);
            if (k0 + 32 < K)                 // next-next slab -> global_prefetch
                __builtin_prefetch(wp + (size_t)32 * N, 0, 1);
        }
    };

    // ---- prologue: fill buffer 0 ------------------------------------------
    loadA(0, 0);
    loadB(0, 0);
    wait_asynccnt0();
    __syncthreads();

    int cur = 0;
    for (int k0 = 0; k0 < K; k0 += 32, cur ^= 1) {
        const int  nxt      = cur ^ 1;
        const bool has_next = (k0 + 32 < K);

        // kick async copies for next tile; they run under the WMMAs below
        if (has_next) { loadA(nxt, k0 + 32); loadB(nxt, k0 + 32); }

        // ---- fragments per ISA 16-bit layouts -----------------------------
        ABfrag a[4], b[2];
        const int mrow = lane & 15;
        const int kg   = (lane >> 4) << 3;   // 0 or 8
#pragma unroll
        for (int ms = 0; ms < 4; ++ms) {
            const unsigned short* p = &sA[cur][(wm + ms * 16 + mrow) * 32];
            a[ms].u[0] = *(const u32x4*)&p[kg];        // K = kg..kg+7
            a[ms].u[1] = *(const u32x4*)&p[16 + kg];   // K = 16+kg..16+kg+7
        }
#pragma unroll
        for (int ns = 0; ns < 2; ++ns) {
            const unsigned short* p = &sB[cur][wn + ns * 16 + mrow];
#pragma unroll
            for (int j = 0; j < 4; ++j) {
                b[ns].u[0][j] = (unsigned)p[(kg + 2 * j) * 128] | ((unsigned)p[(kg + 2 * j + 1) * 128] << 16);
                b[ns].u[1][j] = (unsigned)p[(16 + kg + 2 * j) * 128] | ((unsigned)p[(16 + kg + 2 * j + 1) * 128] << 16);
            }
        }
#pragma unroll
        for (int ms = 0; ms < 4; ++ms)
#pragma unroll
            for (int ns = 0; ns < 2; ++ns)
                acc[ms][ns] = __builtin_amdgcn_wmma_f32_16x16x32_bf16(
                    false, a[ms].v, false, b[ns].v,
                    (short)0, acc[ms][ns], false, false);

        wait_asynccnt0();
        __syncthreads();
    }

    // ---- epilogue: C/D layout lane[0..15]=N, VGPR v -> M=v (+8 hi lanes) ---
    const int ncol = lane & 15;
    const int msub = (lane >> 4) << 3;
#pragma unroll
    for (int ms = 0; ms < 4; ++ms)
#pragma unroll
        for (int ns = 0; ns < 2; ++ns) {
            const int col = bn + wn + ns * 16 + ncol;
            const float bv = bias[col];
#pragma unroll
            for (int v = 0; v < 8; ++v) {
                const int row = bm + wm + ms * 16 + msub + v;
                float val = acc[ms][ns][v] + bv;
                if (MODE == 0) {
                    val = apply_act<ACT>(val);
                    ((unsigned short*)Out)[(size_t)row * N + col] = f2bf(val);
                } else {
                    float* o = (float*)Out + (size_t)row * N + col;
                    if (initOut) *o = val; else *o += val;
                }
            }
        }
}

// ---------------------------------------------------------------------------
// f32 -> bf16 bulk convert (h and weight panels)
// ---------------------------------------------------------------------------
__global__ __launch_bounds__(256)
void f32_to_bf16_kernel(const float* __restrict__ in,
                        unsigned short* __restrict__ out, size_t n)
{
    size_t i = ((size_t)blockIdx.x * blockDim.x + threadIdx.x) * 4;
    if (i + 3 < n) {
        f32x4 v = *(const f32x4*)&in[i];
        u32x2 p;
        p.x = pack2bf(v.x, v.y);
        p.y = pack2bf(v.z, v.w);
        *(u32x2*)&out[i] = p;
    }
}

// ---------------------------------------------------------------------------
// aux head: out[m,0..3] = X[m,:](bf16) @ Wc[:,0..3] + bc  (one wave per row)
// ---------------------------------------------------------------------------
__global__ __launch_bounds__(256)
void aux_head_kernel(const unsigned short* __restrict__ X,
                     const float* __restrict__ Wc,
                     const float* __restrict__ bc,
                     float* __restrict__ out, int M, int K)
{
    const int wrow = (int)((blockIdx.x * (size_t)blockDim.x + threadIdx.x) >> 5);
    const int lane = threadIdx.x & 31;
    if (wrow >= M) return;
    const unsigned short* xr = X + (size_t)wrow * K;
    float a0 = 0.f, a1 = 0.f, a2 = 0.f, a3 = 0.f;
    for (int k = lane; k < K; k += 32) {
        float xv = bf2f(xr[k]);
        f32x4 w = *(const f32x4*)&Wc[(size_t)k * 4];
        a0 += xv * w.x; a1 += xv * w.y; a2 += xv * w.z; a3 += xv * w.w;
    }
#pragma unroll
    for (int off = 16; off > 0; off >>= 1) {
        a0 += __shfl_xor(a0, off, 32);
        a1 += __shfl_xor(a1, off, 32);
        a2 += __shfl_xor(a2, off, 32);
        a3 += __shfl_xor(a3, off, 32);
    }
    if (lane == 0) {
        float* o = out + (size_t)wrow * 4;
        o[0] = a0 + bc[0]; o[1] = a1 + bc[1];
        o[2] = a2 + bc[2]; o[3] = a3 + bc[3];
    }
}

// ---------------------------------------------------------------------------
// host-side dispatch
// ---------------------------------------------------------------------------
static void launch_convert(const float* in, unsigned short* out, size_t n,
                           hipStream_t s)
{
    f32_to_bf16_kernel<<<(unsigned)(n / 4 / 256), 256, 0, s>>>(in, out, n);
}

static void launch_gemm_bf16(int act, const unsigned short* A,
                             const unsigned short* Wb, const float* bias,
                             unsigned short* Out, int M, int N, int K,
                             hipStream_t s)
{
    dim3 grid(N / 128, M / 128), blk(256);
    switch (act) {
    case 0: gemm_bias_act<0, 0><<<grid, blk, 0, s>>>(A, Wb, bias, Out, M, N, K, 0); break;
    case 1: gemm_bias_act<1, 0><<<grid, blk, 0, s>>>(A, Wb, bias, Out, M, N, K, 0); break;
    case 2: gemm_bias_act<2, 0><<<grid, blk, 0, s>>>(A, Wb, bias, Out, M, N, K, 0); break;
    default: gemm_bias_act<3, 0><<<grid, blk, 0, s>>>(A, Wb, bias, Out, M, N, K, 0); break;
    }
}

extern "C" void kernel_launch(void* const* d_in, const int* in_sizes, int n_in,
                              void* d_out, int out_size, void* d_ws, size_t ws_size,
                              hipStream_t stream)
{
    (void)in_sizes; (void)n_in; (void)out_size; (void)ws_size;

    const int E = 6, D = 1024, H = 4096, OUT = 4;
    const int BT = 4 * 2048;  // 8192
    static const int act_ids[6] = {0, 1, 2, 3, 0, 1}; // relu,gelu,tanh,silu,relu,gelu

    const float* h  = (const float*)d_in[0];
    const float* W1 = (const float*)d_in[1];
    const float* b1 = (const float*)d_in[2];
    const float* W2 = (const float*)d_in[3];
    const float* b2 = (const float*)d_in[4];
    const float* Wm = (const float*)d_in[5];
    const float* bm = (const float*)d_in[6];
    const float* Wc = (const float*)d_in[7];
    const float* bc = (const float*)d_in[8];

    // workspace (bf16): hb 16MB | x1 64MB | x2 64MB | wb1 8MB | wb2 32MB | wbm 8MB
    unsigned short* hb  = (unsigned short*)d_ws;
    unsigned short* x1  = hb  + (size_t)BT * D;
    unsigned short* x2  = x1  + (size_t)BT * H;
    unsigned short* wb1 = x2  + (size_t)BT * H;
    unsigned short* wb2 = wb1 + (size_t)D * H;
    unsigned short* wbm = wb2 + (size_t)H * H;

    float* mainOut = (float*)d_out;                      // [BT, D]
    float* auxOut  = mainOut + (size_t)BT * D;           // [E, BT, OUT]

    // h -> bf16 (once)
    launch_convert(h, hb, (size_t)BT * D, stream);

    for (int e = 0; e < E; ++e) {
        const int act = act_ids[e];

        // stage this expert's weights in bf16 (streaming pass)
        launch_convert(W1 + (size_t)e * D * H, wb1, (size_t)D * H, stream);
        launch_convert(W2 + (size_t)e * H * H, wb2, (size_t)H * H, stream);
        launch_convert(Wm + (size_t)e * H * D, wbm, (size_t)H * D, stream);

        // x1 = act(h @ W1[e] + b1[e])          [8192,4096]
        launch_gemm_bf16(act, hb, wb1, b1 + (size_t)e * H, x1, BT, H, D, stream);
        // x2 = act(x1 @ W2[e] + b2[e])         [8192,4096]
        launch_gemm_bf16(act, x1, wb2, b2 + (size_t)e * H, x2, BT, H, H, stream);
        // main (+)= x2 @ Wm[e] + bm[e]         [8192,1024] f32
        {
            dim3 grid(D / 128, BT / 128), blk(256);
            gemm_bias_act<4, 1><<<grid, blk, 0, stream>>>(
                x2, wbm, bm + (size_t)e * D,
                (void*)mainOut, BT, D, H, (e == 0) ? 1 : 0);
        }
        // aux[e] = x2 @ Wc[e] + bc[e]          [8192,4]
        {
            int blocks = (BT * 32 + 255) / 256;   // one wave per row
            aux_head_kernel<<<blocks, 256, 0, stream>>>(
                x2, Wc + (size_t)e * H * OUT, bc + (size_t)e * OUT,
                auxOut + (size_t)e * BT * OUT, BT, H);
        }
    }
}
